// GraphAttentionMLP_21139829030951
// MI455X (gfx1250) — matrix operand order repaired and checked by reading, not compiled, not run on hardware
//
#include <hip/hip_runtime.h>

typedef __attribute__((ext_vector_type(16))) _Float16 v16h;
typedef __attribute__((ext_vector_type(8)))  float    v8f;
typedef _Float16 half_t;

#define E_   320000
#define N_   20000
#define C_   128
#define A_   4
#define S_   16
#define H_   8
#define HD_  32
#define AH_  256
#define RH_  64
#define OUT_ 128

// ---------------- WMMA fragment helpers (CDNA5 wave32 layouts) ----------------

// A fragment: 16x32 f16, row-major LDS buffer [16][lda], rows M=0..15, cols K=k0..k0+31
// lane L: M = L&15, half = L>>4. VGPR v holds K = k0 + (v<4 ? 2v : 2v+8) + 8*half.
__device__ __forceinline__ v16h load_A(const half_t* __restrict__ buf, int lda,
                                       int k0, int lane) {
  const int m  = lane & 15;
  const int hf = lane >> 4;
  const half_t* row = buf + m * lda + k0 + 8 * hf;
  v16h r;
#pragma unroll
  for (int v = 0; v < 8; ++v) {
    const int ko = (v < 4) ? (2 * v) : (2 * v + 8);
    r[2 * v]     = row[ko];
    r[2 * v + 1] = row[ko + 1];
  }
  return r;
}

// B fragment: 32x16 f16 from pair-interleaved weights Wp[(k>>1)*NC + n]*2 + (k&1).
// lane L: N = n0 + (L&15), half = L>>4. VGPR v holds K = k0 + (v<4 ? 2v : 2v+8) + 8*half.
template <int NC>
__device__ __forceinline__ v16h load_B(const half_t* __restrict__ wp,
                                       int k0, int n0, int lane) {
  const int n  = n0 + (lane & 15);
  const int hf = lane >> 4;
  v16h r;
#pragma unroll
  for (int v = 0; v < 8; ++v) {
    const int k2 = (k0 >> 1) + ((v < 4) ? v : (v + 4)) + 4 * hf;
    const half_t* p = wp + (((k2 * NC) + n) << 1);
    r[2 * v]     = p[0];
    r[2 * v + 1] = p[1];
  }
  return r;
}

template <int K, int NC>
__device__ __forceinline__ v8f gemm_frag(const half_t* __restrict__ Abuf, int lda,
                                         const half_t* __restrict__ Bp,
                                         int n0, int lane) {
  v8f c;
#pragma unroll
  for (int r = 0; r < 8; ++r) c[r] = 0.0f;
#pragma unroll
  for (int k0 = 0; k0 < K; k0 += 32) {
    const v16h a = load_A(Abuf, lda, k0, lane);
    const v16h b = load_B<NC>(Bp, k0, n0, lane);
    c = __builtin_amdgcn_wmma_f32_16x16x32_f16(false, a, false, b,
                                               (short)0, c, false, false);
  }
  return c;
}

// pair-interleave f32 weight [K][NC] into f16 LDS
template <int K, int NC>
__device__ __forceinline__ void load_pairs(half_t* __restrict__ dst,
                                           const float* __restrict__ src, int tid) {
  for (int i = tid; i < K * NC; i += 256) {
    const int k = i / NC, n = i - k * NC;
    dst[(((k >> 1) * NC + n) << 1) + (k & 1)] = (half_t)src[i];
  }
}

// fast sigmoid / silu via v_rcp_f32 (WMMA inputs are f16 anyway; no need for
// the IEEE div_scale/div_fixup expansion the compiler emits for '/')
__device__ __forceinline__ float sigmoid_f(float x) {
  return __builtin_amdgcn_rcpf(1.0f + __expf(-x));
}
__device__ __forceinline__ float silu_f(float x) {
  return x * sigmoid_f(x);
}

// ---------------- Kernel 0: init scratch ----------------
__global__ void init_kernel(float* __restrict__ amax, float* __restrict__ denom,
                            float* __restrict__ node_acc) {
  const int i = blockIdx.x * 256 + threadIdx.x;
  if (i < N_ * H_) { amax[i] = -__builtin_inff(); denom[i] = 0.0f; }
  if (i < N_ * AH_) node_acc[i] = 0.0f;
}

// ---------------- Kernel A: fused per-edge pipeline ----------------
// LDS float section: 16384 floats (65536 B), then f16 section (244992 B)
#define SMEM_A (65536 + 244992)

__global__ void __launch_bounds__(256, 1)
edge_kernel(const float* __restrict__ message, const float* __restrict__ edge_attr,
            const float* __restrict__ edge_scalars,
            const float* __restrict__ w_rad1, const float* __restrict__ b_rad1,
            const float* __restrict__ w_rad2, const float* __restrict__ b_rad2,
            const float* __restrict__ w_dtp_v,
            const float* __restrict__ w_lin_act, const float* __restrict__ b_lin_act,
            const float* __restrict__ w_alpha, const float* __restrict__ b_alpha,
            const float* __restrict__ w_lin_v, const float* __restrict__ b_lin_v,
            const float* __restrict__ alpha_dot, const int* __restrict__ edge_dst,
            half_t* __restrict__ d_value, float* __restrict__ d_logit,
            float* __restrict__ d_amax) {
  extern __shared__ char smem[];
  // f32 section
  float* s_br1  = (float*)smem;          // 64
  float* s_br2  = s_br1 + 64;            // 512
  float* s_bal  = s_br2 + 512;           // 256
  float* s_bla  = s_bal + 256;           // 128
  float* s_blv  = s_bla + 128;           // 256
  float* s_wdtp = s_blv + 256;           // 512  (w_dtp_v [128][4])
  float* s_adot = s_wdtp + 512;          // 256  (alpha_dot [8][32])
  float* s_attr = s_adot + 256;          // 64   ([16][4])
  float* s_wv   = s_attr + 64;           // 8192 ([16][512])
  float* s_alph = s_wv + 8192;           // 4096 ([16][256])
  float* s_act  = s_alph + 4096;         // 2048 ([16][128])
  // f16 section (65536 B offset)
  half_t* s_wr1 = (half_t*)(smem + 65536); // 2048  (K padded to 32, N=64)
  half_t* s_wr2 = s_wr1 + 2048;            // 32768 (64x512)
  half_t* s_wal = s_wr2 + 32768;           // 32768 (128x256)
  half_t* s_wla = s_wal + 32768;           // 16384 (128x128)
  half_t* s_wlv = s_wla + 16384;           // 32768 (128x256)
  half_t* s_es  = s_wlv + 32768;           // 16*34  A-staging
  half_t* s_h   = s_es + 544;              // 16*66
  half_t* s_msg = s_h + 1056;              // 16*130
  half_t* s_val = s_msg + 2080;            // 16*130

  const int tid  = threadIdx.x;
  const int lane = tid & 31;
  const int wv   = tid >> 5;

  // ---- resident weights ----
  for (int i = tid; i < 32 * 64; i += 256) {  // w_rad1 padded K 16->32
    const int k = i >> 6, n = i & 63;
    const float v = (k < S_) ? w_rad1[k * 64 + n] : 0.0f;
    s_wr1[(((k >> 1) * 64 + n) << 1) + (k & 1)] = (half_t)v;
  }
  load_pairs<RH_, C_ * A_>(s_wr2, w_rad2, tid);
  load_pairs<C_, AH_>(s_wal, w_alpha, tid);
  load_pairs<C_, C_>(s_wla, w_lin_act, tid);
  load_pairs<C_, AH_>(s_wlv, w_lin_v, tid);
  for (int i = tid; i < 64; i += 256)  s_br1[i] = b_rad1[i];
  for (int i = tid; i < 512; i += 256) s_br2[i] = b_rad2[i];
  for (int i = tid; i < 256; i += 256) s_bal[i] = b_alpha[i];
  if (tid < 128) s_bla[tid] = b_lin_act[tid];
  for (int i = tid; i < 256; i += 256) s_blv[i] = b_lin_v[i];
  for (int i = tid; i < 512; i += 256) s_wdtp[i] = w_dtp_v[i];
  for (int i = tid; i < 256; i += 256) s_adot[i] = alpha_dot[i];
  __syncthreads();

  const int TILES = E_ / 16;
  for (int t = blockIdx.x; t < TILES; t += gridDim.x) {
    const int e0 = t * 16;

    // stage 0: stage edge_scalars (K padded to 32) + edge_attr; prefetch next message
    for (int i = tid; i < 16 * 32; i += 256) {
      const int r = i >> 5, k = i & 31;
      const float v = (k < S_) ? edge_scalars[(size_t)(e0 + r) * S_ + k] : 0.0f;
      s_es[r * 34 + k] = (half_t)v;
    }
    if (tid < 64) s_attr[tid] = edge_attr[(size_t)e0 * A_ + tid];
    const int tn = t + gridDim.x;
    if (tn < TILES && tid < 128)
      __builtin_prefetch(message + (size_t)tn * 16 * C_ + tid * 16, 0, 1);
    __syncthreads();

    // stage 1: h = silu(es @ w_rad1 + b_rad1)  [16][64]
    if (wv < 4) {
      const int n0 = wv * 16;
      v8f c = gemm_frag<32, 64>(s_es, 34, s_wr1, n0, lane);
      const float bn = s_br1[n0 + (lane & 15)];
      const int M0 = (lane >> 4) * 8, col = n0 + (lane & 15);
#pragma unroll
      for (int r = 0; r < 8; ++r)
        s_h[(M0 + r) * 66 + col] = (half_t)silu_f(c[r] + bn);
    }
    __syncthreads();

    // stage 2: wvals = h @ w_rad2 + b_rad2  [16][512]
#pragma unroll
    for (int f = wv; f < 32; f += 8) {
      const int n0 = f * 16;
      v8f c = gemm_frag<64, 512>(s_h, 66, s_wr2, n0, lane);
      const float bn = s_br2[n0 + (lane & 15)];
      const int M0 = (lane >> 4) * 8, col = n0 + (lane & 15);
#pragma unroll
      for (int r = 0; r < 8; ++r) s_wv[(M0 + r) * 512 + col] = c[r] + bn;
    }
    __syncthreads();

    // stage 3: msg[e,c] = message[e,c] * sum_a wvals[e,4c+a]*attr[e,a]
    for (int i = tid; i < 16 * C_; i += 256) {
      const int r = i >> 7, c = i & 127;
      const float* wr = s_wv + r * 512 + c * 4;
      const float* at = s_attr + r * 4;
      const float s = wr[0] * at[0] + wr[1] * at[1] + wr[2] * at[2] + wr[3] * at[3];
      s_msg[r * 130 + c] = (half_t)(message[(size_t)(e0 + r) * C_ + c] * s);
    }
    __syncthreads();

    // stage 4: alpha = slr(msg @ w_alpha + b_alpha)  [16][256]
#pragma unroll
    for (int f = wv; f < 16; f += 8) {
      const int n0 = f * 16;
      v8f c = gemm_frag<128, 256>(s_msg, 130, s_wal, n0, lane);
      const float bn = s_bal[n0 + (lane & 15)];
      const int M0 = (lane >> 4) * 8, col = n0 + (lane & 15);
#pragma unroll
      for (int r = 0; r < 8; ++r) {
        const float x = c[r] + bn;
        // smooth leaky relu (a=0.2): x*(0.2 + 0.8*sigmoid(x))
        s_alph[(M0 + r) * 256 + col] = x * (0.2f + 0.8f * sigmoid_f(x));
      }
    }
    __syncthreads();

    // stage 5: logits + scatter max (int-bits float atomic-max trick)
    if (tid < 128) {
      const int r = tid >> 3, hh = tid & 7;
      const float* ar = s_alph + r * 256 + hh * HD_;
      const float* dr = s_adot + hh * HD_;
      float acc = 0.0f;
#pragma unroll
      for (int k = 0; k < HD_; ++k) acc += ar[k] * dr[k];
      const int e = e0 + r;
      d_logit[e * H_ + hh] = acc;
      float* addr = d_amax + (size_t)edge_dst[e] * H_ + hh;
      if (acc >= 0.0f) atomicMax((int*)addr, __float_as_int(acc));
      else             atomicMin((unsigned int*)addr, __float_as_uint(acc));
    }

    // stage 6: act = silu(msg @ w_lin_act + b)  [16][128]
    {
      const int n0 = wv * 16;
      v8f c = gemm_frag<128, 128>(s_msg, 130, s_wla, n0, lane);
      const float bn = s_bla[n0 + (lane & 15)];
      const int M0 = (lane >> 4) * 8, col = n0 + (lane & 15);
#pragma unroll
      for (int r = 0; r < 8; ++r)
        s_act[(M0 + r) * 128 + col] = silu_f(c[r] + bn);
    }
    __syncthreads();

    // stage 7: value_pre[e,c] = act[e,c] * sum_a w_dtp_v[c,a]*attr[e,a]
    for (int i = tid; i < 16 * C_; i += 256) {
      const int r = i >> 7, c = i & 127;
      const float* at = s_attr + r * 4;
      const float* wd = s_wdtp + c * 4;
      const float s = wd[0] * at[0] + wd[1] * at[1] + wd[2] * at[2] + wd[3] * at[3];
      s_val[r * 130 + c] = (half_t)(s_act[r * 128 + c] * s);
    }
    __syncthreads();

    // stage 8: value = value_pre @ w_lin_v + b_lin_v -> global [E][256] f16
#pragma unroll
    for (int f = wv; f < 16; f += 8) {
      const int n0 = f * 16;
      v8f c = gemm_frag<128, 256>(s_val, 130, s_wlv, n0, lane);
      const float bn = s_blv[n0 + (lane & 15)];
      const int M0 = (lane >> 4) * 8, col = n0 + (lane & 15);
#pragma unroll
      for (int r = 0; r < 8; ++r)
        d_value[(size_t)(e0 + M0 + r) * AH_ + col] = (half_t)(c[r] + bn);
    }
    __syncthreads();
  }
}

// ---------------- Kernel B: exp + scatter denom ----------------
__global__ void denom_kernel(const float* __restrict__ d_logit,
                             const int* __restrict__ edge_dst,
                             const float* __restrict__ d_amax,
                             float* __restrict__ d_denom) {
  const int i = blockIdx.x * 256 + threadIdx.x;
  if (i >= E_ * H_) return;
  const int e = i >> 3, hh = i & 7;
  const int dst = edge_dst[e];
  float m = d_amax[(size_t)dst * H_ + hh];
  if (m < -3.0e38f) m = 0.0f;  // isfinite fixup (matches reference)
  atomicAdd(d_denom + (size_t)dst * H_ + hh, __expf(d_logit[i] - m));
}

// ---------------- Kernel C: normalized weighted scatter-add ----------------
__global__ void aggregate_kernel(const half_t* __restrict__ d_value,
                                 const float* __restrict__ d_logit,
                                 const int* __restrict__ edge_dst,
                                 const float* __restrict__ d_amax,
                                 const float* __restrict__ d_denom,
                                 float* __restrict__ node_acc) {
  const int i = blockIdx.x * 256 + threadIdx.x;  // E_*AH_ = 81,920,000 < 2^31
  const int e = i >> 8, c = i & 255, hh = c >> 5;
  const int dst = edge_dst[e];
  float m = d_amax[(size_t)dst * H_ + hh];
  if (m < -3.0e38f) m = 0.0f;
  const float coeff = __expf(d_logit[e * H_ + hh] - m) *
                      __builtin_amdgcn_rcpf(d_denom[(size_t)dst * H_ + hh] + 1e-16f);
  atomicAdd(node_acc + (size_t)dst * AH_ + c, (float)d_value[i] * coeff);
}

// ---------------- Kernel D: node projection (WMMA) ----------------
#define SMEM_D (512 + 65536 + 16 * 258 * 2)

__global__ void __launch_bounds__(256)
proj_kernel(const float* __restrict__ node_acc, const float* __restrict__ w_proj,
            const float* __restrict__ b_proj, float* __restrict__ out) {
  extern __shared__ char smem[];
  float*  s_bp = (float*)smem;                 // 128 f32
  half_t* s_wp = (half_t*)(smem + 512);        // 256x128 pair-interleaved
  half_t* s_nd = s_wp + 32768;                 // [16][258] A-staging

  const int tid  = threadIdx.x;
  const int lane = tid & 31;
  const int wv   = tid >> 5;

  load_pairs<AH_, OUT_>(s_wp, w_proj, tid);
  if (tid < 128) s_bp[tid] = b_proj[tid];
  __syncthreads();

  const int TILES = N_ / 16;
  for (int t = blockIdx.x; t < TILES; t += gridDim.x) {
    const int r0 = t * 16;
    for (int i = tid; i < 16 * AH_; i += 256) {
      const int r = i >> 8, k = i & 255;
      s_nd[r * 258 + k] = (half_t)node_acc[(size_t)(r0 + r) * AH_ + k];
    }
    __syncthreads();
    {
      const int n0 = wv * 16;  // 8 waves x 16 cols = OUT_
      v8f c = gemm_frag<256, 128>(s_nd, 258, s_wp, n0, lane);
      const float bn = s_bp[n0 + (lane & 15)];
      const int M0 = (lane >> 4) * 8, col = n0 + (lane & 15);
#pragma unroll
      for (int r = 0; r < 8; ++r)
        out[(size_t)(r0 + M0 + r) * OUT_ + col] = c[r] + bn;
    }
    __syncthreads();
  }
}

// ---------------- Host launcher ----------------
extern "C" void kernel_launch(void* const* d_in, const int* in_sizes, int n_in,
                              void* d_out, int out_size, void* d_ws, size_t ws_size,
                              hipStream_t stream) {
  const float* message      = (const float*)d_in[0];
  const float* edge_attr    = (const float*)d_in[1];
  const float* edge_scalars = (const float*)d_in[2];
  const float* w_rad1       = (const float*)d_in[3];
  const float* b_rad1       = (const float*)d_in[4];
  const float* w_rad2       = (const float*)d_in[5];
  const float* b_rad2       = (const float*)d_in[6];
  const float* w_dtp_v      = (const float*)d_in[7];
  const float* w_lin_act    = (const float*)d_in[8];
  const float* b_lin_act    = (const float*)d_in[9];
  const float* w_alpha      = (const float*)d_in[10];
  const float* b_alpha      = (const float*)d_in[11];
  const float* w_lin_v      = (const float*)d_in[12];
  const float* b_lin_v      = (const float*)d_in[13];
  const float* alpha_dot    = (const float*)d_in[14];
  const float* w_proj       = (const float*)d_in[15];
  const float* b_proj       = (const float*)d_in[16];
  const int*   edge_dst     = (const int*)d_in[17];
  (void)in_sizes; (void)n_in; (void)out_size; (void)ws_size;

  // workspace carve-out
  char* ws = (char*)d_ws;
  half_t* d_value = (half_t*)ws;                 size_t off = (size_t)E_ * AH_ * 2;
  float* d_logit  = (float*)(ws + off);          off += (size_t)E_ * H_ * 4;
  float* d_amax   = (float*)(ws + off);          off += (size_t)N_ * H_ * 4;
  float* d_denom  = (float*)(ws + off);          off += (size_t)N_ * H_ * 4;
  float* d_node   = (float*)(ws + off);

  (void)hipFuncSetAttribute((const void*)edge_kernel,
                            hipFuncAttributeMaxDynamicSharedMemorySize, SMEM_A);
  (void)hipFuncSetAttribute((const void*)proj_kernel,
                            hipFuncAttributeMaxDynamicSharedMemorySize, SMEM_D);

  init_kernel<<<(N_ * AH_ + 255) / 256, 256, 0, stream>>>(d_amax, d_denom, d_node);

  edge_kernel<<<640, 256, SMEM_A, stream>>>(
      message, edge_attr, edge_scalars, w_rad1, b_rad1, w_rad2, b_rad2,
      w_dtp_v, w_lin_act, b_lin_act, w_alpha, b_alpha, w_lin_v, b_lin_v,
      alpha_dot, edge_dst, d_value, d_logit, d_amax);

  denom_kernel<<<(E_ * H_ + 255) / 256, 256, 0, stream>>>(
      d_logit, edge_dst, d_amax, d_denom);

  aggregate_kernel<<<(E_ * AH_) / 256, 256, 0, stream>>>(
      d_value, d_logit, edge_dst, d_amax, d_denom, d_node);

  proj_kernel<<<160, 256, SMEM_D, stream>>>(d_node, w_proj, b_proj, (float*)d_out);
}
